// PolyphonicDDSPPiano_8057358648356
// MI455X (gfx1250) — matrix operand order repaired and checked by reading, not compiled, not run on hardware
//
#include <hip/hip_runtime.h>
#include <hip/hip_bf16.h>
#include <math.h>

#define SRATE   16000
#define NSAMP   48000
#define TFRM    750
#define NHARM   96
#define IRLEN   24000
#define NB      6
#define NVOICE  2
#define NROWS   9000   // NVOICE * NB * TFRM

// ---- workspace layout (floats) ----
#define O_CIN   0L           // 9000*34   = 306000
#define O_CT    306000L      // 6*750*32  = 144000
#define O_H0    450000L      // 9000*128  = 1152000
#define O_XG    1602000L     // 9000*576  = 5184000
#define O_H     6786000L     // 9000*192  = 1728000
#define O_H2    8514000L     // 9000*192  = 1728000
#define O_AMP   10242000L    // 9000
#define O_HARM  10251000L    // 9000*96   = 864000
#define O_NOI   11115000L    // 9000*64   = 576000
#define O_AUD   11691000L    // 6*48000   = 288000
// total 11,979,000 floats ~ 47.9 MB

typedef __attribute__((ext_vector_type(16))) __bf16 v16bf;
typedef __attribute__((ext_vector_type(8)))  float  v8f;

static __device__ __forceinline__ __bf16 bf_from_bits(unsigned short u) {
    __bf16 h; __builtin_memcpy(&h, &u, 2); return h;
}
static __device__ __forceinline__ __bf16 f2bf(float f) {
    unsigned x = __float_as_uint(f);
    unsigned r = x + 0x7fffu + ((x >> 16) & 1u);
    return bf_from_bits((unsigned short)(r >> 16));
}
static __device__ __forceinline__ float sigm(float x) { return 1.f / (1.f + __expf(-x)); }
static __device__ __forceinline__ float msigf(float x) {
    float s = sigm(x); return 2.f * __powf(s, 2.302585093f) + 1e-7f;
}
static __device__ __forceinline__ int imin(int a, int b) { return a < b ? a : b; }
static __device__ __forceinline__ int imax(int a, int b) { return a > b ? a : b; }

// ---------------- WMMA GEMM, bf16 hi/lo split (~f32 precision), f32 accum ----------------
// C[M,N] = act(A[M,K] @ B[K,N] + bias).  N must be a multiple of 16.
__global__ void gemm_wmma_bf16x2(const float* __restrict__ A, int lda,
                                 const float* __restrict__ B, int ldb,
                                 const float* __restrict__ bias,
                                 float* __restrict__ C, int ldc,
                                 int M, int N, int K, int act)
{
    const int lane = threadIdx.x & 31;
    const int wave = threadIdx.x >> 5;
    const int half = lane >> 4;
    const int lrow = lane & 15;
    const int ntile = N >> 4;
    const int mtile = (M + 15) >> 4;
    int tile = blockIdx.x * (blockDim.x >> 5) + wave;
    if (tile >= ntile * mtile) return;
    int mt = tile / ntile, nt = tile - mt * ntile;
    int arow = mt * 16 + lrow;                  // may exceed M-1 in last m-tile
    int bcol = nt * 16 + lrow;                  // always < N
    const bool mfull = (mt * 16 + 15) < M;      // wave-uniform
    const float* Arow = A + (long)imin(arow, M - 1) * lda;
    const bool rowok = arow < M;

    v8f acc = {};
    v8f accl = {};
    for (int k0 = 0; k0 < K; k0 += 32) {
        v16bf ah, al, bh, bl;
        if (mfull & ((k0 + 31) < K)) {
            // interior: unconditional loads, no masking at all
            __builtin_prefetch(Arow + k0 + 64, 0, 1);
#pragma unroll
            for (int e = 0; e < 16; ++e) {
                int ka = half * 16 + e;   // A 16x32 bf16 layout
                float av = Arow[k0 + ka];
                unsigned xa = __float_as_uint(av);
                unsigned hba = (xa + 0x8000u) & 0xffff0000u;
                ah[e] = bf_from_bits((unsigned short)(hba >> 16));
                al[e] = f2bf(av - __uint_as_float(hba));
                int kb = half * 16 + e;                          // B 32x16 bf16 layout
                float bv = B[(long)(k0 + kb) * ldb + bcol];
                unsigned xb = __float_as_uint(bv);
                unsigned hbb = (xb + 0x8000u) & 0xffff0000u;
                bh[e] = bf_from_bits((unsigned short)(hbb >> 16));
                bl[e] = f2bf(bv - __uint_as_float(hbb));
            }
        } else {
            // edge: clamp addresses (loads stay unconditional), mask values
#pragma unroll
            for (int e = 0; e < 16; ++e) {
                int ka = half * 16 + e;
                int kk = k0 + ka;
                float av = Arow[imin(kk, K - 1)];
                av = (rowok && kk < K) ? av : 0.f;
                unsigned xa = __float_as_uint(av);
                unsigned hba = (xa + 0x8000u) & 0xffff0000u;
                ah[e] = bf_from_bits((unsigned short)(hba >> 16));
                al[e] = f2bf(av - __uint_as_float(hba));
                int k2 = k0 + half * 16 + e;
                float bv = B[(long)imin(k2, K - 1) * ldb + bcol];
                bv = (k2 < K) ? bv : 0.f;
                unsigned xb = __float_as_uint(bv);
                unsigned hbb = (xb + 0x8000u) & 0xffff0000u;
                bh[e] = bf_from_bits((unsigned short)(hbb >> 16));
                bl[e] = f2bf(bv - __uint_as_float(hbb));
            }
        }
        acc  = __builtin_amdgcn_wmma_f32_16x16x32_bf16(false, ah, false, bh, (short)0, acc,  false, false);
        accl = __builtin_amdgcn_wmma_f32_16x16x32_bf16(false, ah, false, bl, (short)0, accl, false, false);
        accl = __builtin_amdgcn_wmma_f32_16x16x32_bf16(false, al, false, bh, (short)0, accl, false, false);
    }
#pragma unroll
    for (int i = 0; i < 8; ++i) {
        int m = mt * 16 + half * 8 + i;
        if (m < M) {
            float v = acc[i] + accl[i];
            if (bias) v += bias[bcol];
            if (act == 1)      v = (v >= 0.f) ? v : 0.2f * v;       // leaky
            else if (act == 2) v = msigf(v);                        // msig
            C[(long)m * ldc + bcol] = v;
        }
    }
}

// ---------------- context network: MLP + GRU(64) + out, one persistent WG ----------------
__global__ void ctx_forward(const float* __restrict__ pitches, const float* __restrict__ velocities,
                            const float* __restrict__ pedal,
                            const float* __restrict__ Win, const float* __restrict__ bin,
                            const float* __restrict__ gruK, const float* __restrict__ gruR,
                            const float* __restrict__ grub,
                            const float* __restrict__ Wout, const float* __restrict__ bout,
                            float* __restrict__ ct_out)
{
    __shared__ float s_in[NB * 3], s_c[NB * 32], s_xg[NB * 192], s_rg[NB * 192], s_h[NB * 64];
    int tid = threadIdx.x;   // blockDim = 384
    for (int i = tid; i < NB * 64; i += blockDim.x) s_h[i] = 0.f;
    __syncthreads();
    for (int t = 0; t < TFRM; ++t) {
        if (tid < NB) {
            int b = tid;
            float p0 = pitches[(b * 2 + 0) * TFRM + t], p1 = pitches[(b * 2 + 1) * TFRM + t];
            float v0 = velocities[(b * 2 + 0) * TFRM + t], v1 = velocities[(b * 2 + 1) * TFRM + t];
            float a0 = (v0 > 0.f) ? p0 / 127.f : 0.f;
            float a1 = (v1 > 0.f) ? p1 / 127.f : 0.f;
            float cnt = ((v0 > 0.f) ? 1.f : 0.f) + ((v1 > 0.f) ? 1.f : 0.f);
            s_in[b * 3 + 0] = pedal[b * TFRM + t];
            s_in[b * 3 + 1] = v0 + v1;
            s_in[b * 3 + 2] = (a0 + a1) / (cnt + 1e-7f);
        }
        __syncthreads();
        for (int i = tid; i < NB * 32; i += blockDim.x) {
            int b = i >> 5, u = i & 31;
            float a = bin[u];
            for (int j = 0; j < 3; ++j) a += s_in[b * 3 + j] * Win[j * 32 + u];
            s_c[i] = (a >= 0.f) ? a : 0.2f * a;
        }
        __syncthreads();
        for (int i = tid; i < NB * 192; i += blockDim.x) {
            int b = i / 192, g = i % 192;
            const float* cb = s_c + b * 32;
            const float* hb = s_h + b * 64;
            float a = grub[g];
            for (int j = 0; j < 32; ++j) a += cb[j] * gruK[j * 192 + g];
            s_xg[i] = a;
            float r = grub[192 + g];
            for (int j = 0; j < 64; ++j) r += hb[j] * gruR[j * 192 + g];
            s_rg[i] = r;
        }
        __syncthreads();
        for (int i = tid; i < NB * 64; i += blockDim.x) {
            int b = i >> 6, u = i & 63;
            float z  = sigm(s_xg[b * 192 + u]       + s_rg[b * 192 + u]);
            float r  = sigm(s_xg[b * 192 + 64 + u]  + s_rg[b * 192 + 64 + u]);
            float hh = tanhf(s_xg[b * 192 + 128 + u] + r * s_rg[b * 192 + 128 + u]);
            s_h[i] = z * s_h[i] + (1.f - z) * hh;
        }
        __syncthreads();
        for (int i = tid; i < NB * 32; i += blockDim.x) {
            int b = i >> 5, o = i & 31;
            const float* hb = s_h + b * 64;
            float a = bout[o];
            for (int j = 0; j < 64; ++j) a += hb[j] * Wout[j * 32 + o];
            ct_out[(b * TFRM + t) * 32 + o] = a;
        }
        __syncthreads();
    }
}

// ---------------- build core_in rows = [pitch/127, vel, c_t(32)] ----------------
__global__ void build_core_in(const float* __restrict__ pitches, const float* __restrict__ velocities,
                              const float* __restrict__ ct, float* __restrict__ cin)
{
    int idx = blockIdx.x * blockDim.x + threadIdx.x;
    if (idx >= NROWS * 34) return;
    int r = idx / 34, c = idx % 34;
    int v = r / (NB * TFRM), rem = r % (NB * TFRM);
    int b = rem / TFRM, t = rem % TFRM;
    float val;
    if (c == 0)      val = pitches[(b * 2 + v) * TFRM + t] / 127.f;
    else if (c == 1) val = velocities[(b * 2 + v) * TFRM + t];
    else             val = ct[(b * TFRM + t) * 32 + (c - 2)];
    cin[idx] = val;
}

// ---------------- core GRU(192), 12 chains, one persistent WG ----------------
__global__ void core_gru(const float* __restrict__ xg, const float* __restrict__ R,
                         const float* __restrict__ grub, float* __restrict__ hout)
{
    __shared__ float s_h[12 * 192];
    __shared__ float s_rg[12 * 576];
    int tid = threadIdx.x;   // blockDim = 768
    for (int i = tid; i < 12 * 192; i += blockDim.x) s_h[i] = 0.f;
    // warm the recurrent matrix into cache (one-shot, 442 KB -> L2)
    for (int i = tid; i < 192 * 576; i += blockDim.x * 16) __builtin_prefetch(R + i, 0, 2);
    __syncthreads();
    for (int t = 0; t < TFRM; ++t) {
        for (int i = tid; i < 12 * 576; i += blockDim.x) {
            int c = i / 576, j = i % 576;
            const float* hb = s_h + c * 192;
            float a = grub[576 + j];
            for (int k = 0; k < 192; ++k) a += hb[k] * R[k * 576 + j];
            s_rg[i] = a;
        }
        __syncthreads();
        for (int i = tid; i < 12 * 192; i += blockDim.x) {
            int c = i / 192, u = i % 192;
            int v = c / NB, b = c % NB;
            long r = (long)(v * NB * TFRM + b * TFRM + t);
            const float* xrow = xg + r * 576;
            float z  = sigm(xrow[u]        + s_rg[c * 576 + u]);
            float rr = sigm(xrow[192 + u]  + s_rg[c * 576 + 192 + u]);
            float hh = tanhf(xrow[384 + u] + rr * s_rg[c * 576 + 384 + u]);
            float hn = z * s_h[i] + (1.f - z) * hh;
            s_h[i] = hn;
            hout[r * 192 + u] = hn;
        }
        __syncthreads();
    }
}

// ---------------- amp head + harmonic-distribution normalization ----------------
__global__ void heads_post(const float* __restrict__ h2, const float* __restrict__ ampW,
                           const float* __restrict__ ampB, float* __restrict__ amp,
                           float* __restrict__ harm)
{
    int r = blockIdx.x * blockDim.x + threadIdx.x;
    if (r >= NROWS) return;
    const float* hb = h2 + (long)r * 192;
    float a = ampB[0];
    for (int k = 0; k < 192; ++k) a += hb[k] * ampW[k];
    amp[r] = msigf(a);
    float* hr = harm + (long)r * 96;
    float s = 0.f;
    for (int j = 0; j < 96; ++j) s += hr[j];
    float inv = 1.f / s;
    for (int j = 0; j < 96; ++j) hr[j] *= inv;
}

__global__ void zero_buf(float* __restrict__ p, int n) {
    int i = blockIdx.x * blockDim.x + threadIdx.x;
    if (i < n) p[i] = 0.f;
}

// ---------------- harmonic synthesis: 24 WGs = (det, voice, batch); phase scan in regs ----------------
__global__ void harm_synth_k(const float* __restrict__ pitches,
                             const float* __restrict__ amp, const float* __restrict__ harm,
                             const float* aB, const float* bB, const float* aT, const float* bT,
                             const float* detW, const float* detb, const float* sdet,
                             float* __restrict__ audio)
{
    __shared__ float buf[256];
    int wg = blockIdx.x;                    // 0..23
    int det = wg / 12, rem = wg % 12;
    int v = rem / NB, b = rem % NB;
    int tid = threadIdx.x;                  // blockDim = 128
    int h = tid;
    float alphaB = aB[0], betaB = bB[0], alphaT = aT[0], betaT = bT[0];
    float dW = detW[0], db = detb[0], sd = tanhf(sdet[0]);
    int rbase = v * NB * TFRM + b * TFRM;
    int pbase = (b * 2 + v) * TFRM;
    float kf = (float)(h + 1);
    const float w0 = 6.28318530717958647f / (float)SRATE;

    auto framef = [&](int t) -> float {
        float p = pitches[pbase + t];
        float f0 = (p > 0.f) ? 440.f * exp2f((p - 69.f) * (1.f / 12.f)) : 0.f;
        float ps = (p > 0.f) ? p : 1.f;
        float Bf = __expf(alphaT * ps + betaT) + __expf(alphaB * ps + betaB);
        float f = f0 * kf * sqrtf(1.f + Bf * kf * kf);
        if (det) {
            float delta = tanhf((p * (1.f / 127.f)) * dW + db) + sd;
            f *= exp2f(delta * (1.f / 12.f));
        }
        return f;
    };

    float phase = 0.f;
    int tcur = -1;
    float fA = 0.f, fB2 = 0.f, hA = 0.f, hB2 = 0.f;
    for (int n0 = 0; n0 < NSAMP; n0 += 256) {
        buf[tid] = 0.f; buf[tid + 128] = 0.f;
        __syncthreads();
        if (h < NHARM) {
            for (int n = n0; n < n0 + 256; ++n) {
                float src = n * (1.f / 64.f) - 0.4921875f;   // (n+0.5)/64 - 0.5
                float ff = floorf(src);
                int t0 = (int)ff;
                float fr = src - ff;
                if (t0 < 0)        { t0 = 0;        fr = 0.f; }
                if (t0 > TFRM - 2) { t0 = TFRM - 2; fr = 1.f; }
                if (t0 != tcur) {
                    fA = framef(t0); fB2 = framef(t0 + 1);
                    hA  = harm[(long)(rbase + t0)     * 96 + h];
                    hB2 = harm[(long)(rbase + t0 + 1) * 96 + h];
                    tcur = t0;
                }
                float f  = fA + fr * (fB2 - fA);
                float hm = hA + fr * (hB2 - hA);
                phase += f * w0;
                float contrib = (f < 8000.f) ? __sinf(phase) * hm : 0.f;
                atomicAdd(&buf[n - n0], contrib);
            }
        }
        __syncthreads();
        for (int i = tid; i < 256; i += blockDim.x) {
            int n = n0 + i;
            float src = n * (1.f / 64.f) - 0.4921875f;
            float ff = floorf(src);
            int t0 = (int)ff;
            float fr = src - ff;
            if (t0 < 0)        { t0 = 0;        fr = 0.f; }
            if (t0 > TFRM - 2) { t0 = TFRM - 2; fr = 1.f; }
            float a = amp[rbase + t0] * (1.f - fr) + amp[rbase + t0 + 1] * fr;
            atomicAdd(&audio[b * NSAMP + n], buf[i] * a);
        }
        __syncthreads();
    }
}

// ---------------- filtered-noise synthesis: per-frame 128-pt circular conv ----------------
__global__ void noise_synth_k(const float* __restrict__ noisem, const float* __restrict__ white,
                              float* __restrict__ audio)
{
    __shared__ float cosT[128], mags[64], kern[128], frm[128];
    int wg = blockIdx.x;          // 0..11 : v*6+b
    int v = wg / NB, b = wg % NB;
    int tid = threadIdx.x;        // blockDim = 128
    int rbase = v * NB * TFRM + b * TFRM;
    const float* wn = white + ((long)v * NB + b) * NSAMP;
    cosT[tid] = __cosf(6.28318530717958647f * (float)tid * (1.f / 128.f));
    float hann = 0.5f - 0.5f * __cosf(6.28318530717958647f * (float)tid * (1.f / 128.f));
    __syncthreads();
    for (int t = 0; t < TFRM; ++t) {
        if (tid < 64) mags[tid] = noisem[(long)(rbase + t) * 64 + tid];
        int pos = t * 64 + tid;
        frm[tid] = (pos < NSAMP) ? wn[pos] : 0.f;
        __syncthreads();
        {   // kern = irfft(mags padded to 65 bins) : real, via cosine table
            float a = mags[0];
            for (int q = 1; q < 64; ++q) a += 2.f * mags[q] * cosT[(q * tid) & 127];
            kern[tid] = a * (1.f / 128.f);
        }
        __syncthreads();
        {   // filt rolled by WIN/2, windowed, overlap-add
            int n = (tid + 64) & 127;
            float s = 0.f;
            for (int m = 0; m < 128; ++m) s += frm[m] * kern[(n - m) & 127];
            if (pos < NSAMP) atomicAdd(&audio[b * NSAMP + pos], s * hann);
        }
        __syncthreads();
    }
}

// ---------------- reverb: IR convolution as WMMA Toeplitz matmul, bf16 hi/lo split ----------------
__global__ void reverb_wmma(const float* __restrict__ audio, const float* __restrict__ ir,
                            float* __restrict__ out)
{
    const int lane = threadIdx.x & 31;
    const int wave = threadIdx.x >> 5;
    const int half = lane >> 4;
    const int lrow = lane & 15;
    int tile = blockIdx.x * (blockDim.x >> 5) + wave;
    int t0 = tile * 16;
    if (t0 >= NSAMP) return;

    const bool rowok = lrow < NB;
    const float* arow_p = audio + (long)imin(lrow, NB - 1) * NSAMP;

    v8f acc = {}; v8f accl = {};
    int lo = t0 - (IRLEN - 1); if (lo < 0) lo = 0; lo &= ~31;
    int hi = t0 + 15;
    for (int tau0 = lo; tau0 <= hi; tau0 += 32) {
        v16bf ah, al, bh, bl;
#pragma unroll
        for (int e = 0; e < 16; ++e) {
            int ka = half * 16 + e;
            int tau = tau0 + ka;
            float av = arow_p[imin(tau, NSAMP - 1)];          // unconditional load
            av = (rowok && tau < NSAMP) ? av : 0.f;           // branchless mask
            unsigned xa = __float_as_uint(av);
            unsigned hba = (xa + 0x8000u) & 0xffff0000u;
            ah[e] = bf_from_bits((unsigned short)(hba >> 16));
            al[e] = f2bf(av - __uint_as_float(hba));
            int kb = half * 16 + e;
            int d = (t0 + lrow) - (tau0 + kb);                // ir tap for B[k][n]
            float bv = ir[imin(imax(d, 0), IRLEN - 1)];       // unconditional load
            bv = (d >= 0 && d < IRLEN) ? bv : 0.f;            // branchless mask
            unsigned xb = __float_as_uint(bv);
            unsigned hbb = (xb + 0x8000u) & 0xffff0000u;
            bh[e] = bf_from_bits((unsigned short)(hbb >> 16));
            bl[e] = f2bf(bv - __uint_as_float(hbb));
        }
        acc  = __builtin_amdgcn_wmma_f32_16x16x32_bf16(false, ah, false, bh, (short)0, acc,  false, false);
        accl = __builtin_amdgcn_wmma_f32_16x16x32_bf16(false, ah, false, bl, (short)0, accl, false, false);
        accl = __builtin_amdgcn_wmma_f32_16x16x32_bf16(false, al, false, bh, (short)0, accl, false, false);
    }
#pragma unroll
    for (int i = 0; i < 8; ++i) {
        int bb = half * 8 + i;
        int n = t0 + lrow;
        if (bb < NB && n < NSAMP) out[bb * NSAMP + n] = acc[i] + accl[i];
    }
}

// ---------------- host ----------------
static void launch_gemm(const float* A, int lda, const float* B, int ldb, const float* bias,
                        float* C, int ldc, int M, int N, int K, int act, hipStream_t s)
{
    int tiles = ((M + 15) / 16) * (N / 16);
    int blocks = (tiles + 3) / 4;
    gemm_wmma_bf16x2<<<blocks, 128, 0, s>>>(A, lda, B, ldb, bias, C, ldc, M, N, K, act);
}

extern "C" void kernel_launch(void* const* d_in, const int* in_sizes, int n_in,
                              void* d_out, int out_size, void* d_ws, size_t ws_size,
                              hipStream_t stream)
{
    (void)in_sizes; (void)n_in; (void)out_size; (void)ws_size;
    const float* pitches     = (const float*)d_in[0];
    const float* velocities  = (const float*)d_in[1];
    const float* pedal       = (const float*)d_in[2];
    const float* white       = (const float*)d_in[3];
    const float* ctx_Win     = (const float*)d_in[4];
    const float* ctx_bin     = (const float*)d_in[5];
    const float* ctx_gruK    = (const float*)d_in[6];
    const float* ctx_gruR    = (const float*)d_in[7];
    const float* ctx_grub    = (const float*)d_in[8];
    const float* ctx_Wout    = (const float*)d_in[9];
    const float* ctx_bout    = (const float*)d_in[10];
    const float* core_Win    = (const float*)d_in[11];
    const float* core_bin    = (const float*)d_in[12];
    const float* core_gruK   = (const float*)d_in[13];
    const float* core_gruR   = (const float*)d_in[14];
    const float* core_grub   = (const float*)d_in[15];
    const float* core_Wh     = (const float*)d_in[16];
    const float* core_bh     = (const float*)d_in[17];
    const float* amp_W       = (const float*)d_in[18];
    const float* amp_b       = (const float*)d_in[19];
    const float* harm_W      = (const float*)d_in[20];
    const float* harm_b      = (const float*)d_in[21];
    const float* noise_W     = (const float*)d_in[22];
    const float* noise_b     = (const float*)d_in[23];
    const float* alpha_B     = (const float*)d_in[24];
    const float* beta_B      = (const float*)d_in[25];
    const float* alpha_T     = (const float*)d_in[26];
    const float* beta_T      = (const float*)d_in[27];
    const float* det_W       = (const float*)d_in[28];
    const float* det_b       = (const float*)d_in[29];
    const float* static_det  = (const float*)d_in[30];
    const float* ir          = (const float*)d_in[31];
    float* out = (float*)d_out;
    float* w   = (float*)d_ws;

    float* CIN  = w + O_CIN;
    float* CT   = w + O_CT;
    float* H0   = w + O_H0;
    float* XG   = w + O_XG;
    float* H    = w + O_H;
    float* H2   = w + O_H2;
    float* AMP  = w + O_AMP;
    float* HARM = w + O_HARM;
    float* NOI  = w + O_NOI;
    float* AUD  = w + O_AUD;

    zero_buf<<<(NB * NSAMP + 255) / 256, 256, 0, stream>>>(AUD, NB * NSAMP);
    ctx_forward<<<1, 384, 0, stream>>>(pitches, velocities, pedal, ctx_Win, ctx_bin,
                                       ctx_gruK, ctx_gruR, ctx_grub, ctx_Wout, ctx_bout, CT);
    build_core_in<<<(NROWS * 34 + 255) / 256, 256, 0, stream>>>(pitches, velocities, CT, CIN);
    launch_gemm(CIN, 34, core_Win, 128, core_bin, H0, 128, NROWS, 128, 34, 1, stream);
    launch_gemm(H0, 128, core_gruK, 576, core_grub, XG, 576, NROWS, 576, 128, 0, stream);
    core_gru<<<1, 768, 0, stream>>>(XG, core_gruR, core_grub, H);
    launch_gemm(H, 192, core_Wh, 192, core_bh, H2, 192, NROWS, 192, 192, 1, stream);
    launch_gemm(H2, 192, harm_W, 96, harm_b, HARM, 96, NROWS, 96, 192, 2, stream);
    launch_gemm(H2, 192, noise_W, 64, noise_b, NOI, 64, NROWS, 64, 192, 2, stream);
    heads_post<<<(NROWS + 255) / 256, 256, 0, stream>>>(H2, amp_W, amp_b, AMP, HARM);
    harm_synth_k<<<24, 128, 0, stream>>>(pitches, AMP, HARM, alpha_B, beta_B, alpha_T, beta_T,
                                         det_W, det_b, static_det, AUD);
    noise_synth_k<<<12, 128, 0, stream>>>(NOI, white, AUD);
    reverb_wmma<<<750, 128, 0, stream>>>(AUD, ir, out);
}
